// SwinTransformerBlock_30442728194169
// MI455X (gfx1250) — matrix operand order repaired and checked by reading, not compiled, not run on hardware
//
#include <hip/hip_runtime.h>
#include <hip/hip_bf16.h>

// ---------------- static config (mirrors reference) ----------------
#define HH      60
#define WW      60
#define WS7     7
#define SS3     3
#define DIMC    256
#define NHEADS  8
#define HDIM    32
#define NTOK    49            // tokens per window
#define HPAD    63
#define WPADD   63
#define NWIN    81            // windows per image
#define BATCH   24
#define LTOK    (HH*WW)       // 3600
#define NWROWS  (BATCH*NWIN*NTOK)   // 95256 windowed rows
#define MP1     95264               // padded to 16
#define MT1     (MP1/16)            // 5954
#define MTOK    (BATCH*LTOK)        // 86400 token rows
#define MP2     86416               // padded to 16
#define MT2     (MP2/16)            // 5401

typedef __attribute__((ext_vector_type(16))) _Float16 v16h;
typedef __attribute__((ext_vector_type(8)))  _Float16 v8h;
typedef __attribute__((ext_vector_type(8)))  float    v8f;

// ---------------- fragment-layout helpers (CDNA5 ISA) ----------------
// 16-bit A-matrix 16x32: lane l (l<16: K-mid 0..7, l>=16: K-mid 8..15), half i:
//   k = (i<8?0:16) + (l<16?0:8) + (i&7),  m = l&15
// B-matrix 32x16 mirrored: lane ~ column n, same k pattern.
// Packed storage: [tile_r][tile_k][lane 0..31][half 0..15], 1024B per fragment.
__device__ __forceinline__ size_t pk_idx(size_t r, int c, int Ktiles) {
  size_t mt = r >> 4;
  int kt = c >> 5;
  int km = c & 31;
  int ln = (int)(r & 15) + (((km >> 3) & 1) << 4);
  int hh = ((km >> 4) << 3) + (km & 7);
  return ((mt * (size_t)Ktiles + (size_t)kt) * 32 + (size_t)ln) * 16 + (size_t)hh;
}

__device__ __forceinline__ int kform(int hh, int lane) {
  return ((hh >> 3) << 4) + (((lane >> 4) & 1) << 3) + (hh & 7);
}

__device__ __forceinline__ v8f wmma16(v16h a, v16h b, v8f c) {
  return __builtin_amdgcn_wmma_f32_16x16x32_f16(false, a, false, b, (short)0, c,
                                                false, false);
}

// shift-mask region id of a padded-image coordinate (0..62)
__device__ __forceinline__ int sregion(int p) {
  return (p < HPAD - WS7) ? 0 : ((p < HPAD - SS3) ? 1 : 2);
}

// ---------------- weight pack: f32 row-major W[K][N] -> packed-B f16 ----------------
__global__ void pack_w_kernel(const float* __restrict__ W, _Float16* __restrict__ dst,
                              int K, int N) {
  int idx = blockIdx.x * 256 + threadIdx.x;
  if (idx >= K * N) return;
  int k = idx / N, n = idx - k * N;
  dst[pk_idx((size_t)n, k, K >> 5)] = (_Float16)W[idx];
}

// ---------------- LN1 + pad + cyclic shift + window partition -> packed-A ----------
__global__ void ln1_window_pack(const float* __restrict__ x,
                                const float* __restrict__ g,
                                const float* __restrict__ b,
                                _Float16* __restrict__ A1) {
  int lane = threadIdx.x;               // 0..31
  int row  = blockIdx.x * blockDim.y + threadIdx.y;   // windowed row
  if (row >= NWROWS) return;
  int bi  = row / (NWIN * NTOK);
  int rem = row - bi * (NWIN * NTOK);
  int wi  = rem / NTOK;
  int tok = rem - wi * NTOK;
  int i = (wi / 9) * WS7 + tok / WS7;   // position in shifted padded image
  int j = (wi % 9) * WS7 + tok % WS7;
  int h = i + SS3; if (h >= HPAD)  h -= HPAD;   // roll(-3): S[i,j]=P[(i+3)%63,...]
  int w = j + SS3; if (w >= WPADD) w -= WPADD;
  bool valid = (h < HH) && (w < WW);
  if (valid) {
    const float* xp = x + (((size_t)bi * LTOK + (size_t)h * WW + w) * DIMC) + lane * 8;
    float v[8]; float s = 0.f, s2 = 0.f;
#pragma unroll
    for (int u = 0; u < 8; ++u) { v[u] = xp[u]; s += v[u]; s2 += v[u] * v[u]; }
#pragma unroll
    for (int m = 16; m >= 1; m >>= 1) { s += __shfl_xor(s, m, 32); s2 += __shfl_xor(s2, m, 32); }
    float mu = s * (1.f / 256.f);
    float var = s2 * (1.f / 256.f) - mu * mu;
    float rs = rsqrtf(var + 1e-5f);
#pragma unroll
    for (int u = 0; u < 8; ++u) {
      int c = lane * 8 + u;
      float y = (v[u] - mu) * rs * g[c] + b[c];
      A1[pk_idx((size_t)row, c, 8)] = (_Float16)y;
    }
  } else {
#pragma unroll
    for (int u = 0; u < 8; ++u)
      A1[pk_idx((size_t)row, lane * 8 + u, 8)] = (_Float16)0.f;
  }
}

// ---------------- LN2 -> packed-A ----------------
__global__ void ln2_pack(const float* __restrict__ X1,
                         const float* __restrict__ g,
                         const float* __restrict__ b,
                         _Float16* __restrict__ A3) {
  int lane = threadIdx.x;
  int row  = blockIdx.x * blockDim.y + threadIdx.y;   // token row
  if (row >= MTOK) return;
  const float* xp = X1 + (size_t)row * DIMC + lane * 8;
  float v[8]; float s = 0.f, s2 = 0.f;
#pragma unroll
  for (int u = 0; u < 8; ++u) { v[u] = xp[u]; s += v[u]; s2 += v[u] * v[u]; }
#pragma unroll
  for (int m = 16; m >= 1; m >>= 1) { s += __shfl_xor(s, m, 32); s2 += __shfl_xor(s2, m, 32); }
  float mu = s * (1.f / 256.f);
  float var = s2 * (1.f / 256.f) - mu * mu;
  float rs = rsqrtf(var + 1e-5f);
#pragma unroll
  for (int u = 0; u < 8; ++u) {
    int c = lane * 8 + u;
    float y = (v[u] - mu) * rs * g[c] + b[c];
    A3[pk_idx((size_t)row, c, 8)] = (_Float16)y;
  }
}

// ---------------- generic packed-fragment WMMA GEMM ----------------
// MODE 0: qkv    -> f16 row-major [MP1][768], +bias
// MODE 1: proj   -> window-reverse/unshift/crop scatter, +bias, +residual -> X1 f32
// MODE 2: fc1    -> GELU(.+bias) -> packed-A f16 (Ktiles 64)
// MODE 3: fc2    -> X1 + (.+bias) -> d_out f32
template <int MODE>
__global__ __launch_bounds__(128) void gemm_wmma(
    const _Float16* __restrict__ A, const _Float16* __restrict__ Bw,
    const float* __restrict__ bias, int Ktiles,
    _Float16* __restrict__ outH, float* __restrict__ outF,
    const float* __restrict__ resF, int Mvalid) {
  int lane = threadIdx.x & 31;
  int wave = threadIdx.x >> 5;
  int mt = blockIdx.x;
  int ntb = (blockIdx.y * 4 + wave) * 4;      // 4 consecutive N tiles per wave

  const _Float16* Ab = A + ((size_t)mt * Ktiles) * 512 + lane * 16;
  v8f acc[4];
#pragma unroll
  for (int t = 0; t < 4; ++t) acc[t] = (v8f){0.f,0.f,0.f,0.f,0.f,0.f,0.f,0.f};

  for (int kt = 0; kt < Ktiles; ++kt) {
    v16h a = *(const v16h*)(Ab + (size_t)kt * 512);
    if (kt + 1 < Ktiles) __builtin_prefetch(Ab + (size_t)(kt + 1) * 512, 0, 1);
#pragma unroll
    for (int t = 0; t < 4; ++t) {
      const _Float16* Bb = Bw + (((size_t)(ntb + t) * Ktiles + kt) * 512) + lane * 16;
      v16h bfr = *(const v16h*)Bb;
      acc[t] = wmma16(a, bfr, acc[t]);
    }
  }

  int mbase = mt * 16 + ((lane & 16) ? 8 : 0);
#pragma unroll
  for (int t = 0; t < 4; ++t) {
    int n0 = (ntb + t) * 16 + (lane & 15);
    float bn = bias[n0];
#pragma unroll
    for (int r = 0; r < 8; ++r) {
      int row = mbase + r;
      float v = acc[t][r] + bn;
      if (MODE == 0) {
        outH[(size_t)row * 768 + n0] = (_Float16)v;
      } else if (MODE == 1) {
        if (row < Mvalid) {
          int bi  = row / (NWIN * NTOK);
          int rem = row - bi * (NWIN * NTOK);
          int wi  = rem / NTOK;
          int tok = rem - wi * NTOK;
          int i = (wi / 9) * WS7 + tok / WS7;
          int j = (wi % 9) * WS7 + tok % WS7;
          int h = i + SS3; if (h >= HPAD)  h -= HPAD;   // roll(+3) of output
          int w = j + SS3; if (w >= WPADD) w -= WPADD;
          if (h < HH && w < WW) {
            size_t o = ((size_t)bi * LTOK + (size_t)h * WW + w) * DIMC + n0;
            outF[o] = resF[o] + v;
          }
        }
      } else if (MODE == 2) {
        if (row < Mvalid) {
          float ge = 0.5f * v * (1.f + erff(v * 0.70710678118f));  // exact GELU
          outH[pk_idx((size_t)row, n0, 64)] = (_Float16)ge;
        }
      } else { // MODE == 3
        if (row < Mvalid) {
          size_t o = (size_t)row * DIMC + n0;
          outF[o] = resF[o] + v;
        }
      }
    }
  }
}

// ---------------- windowed attention: one wave per (window, head) -------------
__global__ __launch_bounds__(64) void attn_kernel(const _Float16* __restrict__ qkv,
                                                  const float* __restrict__ relTab,
                                                  _Float16* __restrict__ A2) {
  __shared__ __align__(16) _Float16 qs[2][64][32];
  __shared__ __align__(16) _Float16 ks[2][64][32];
  __shared__ __align__(16) _Float16 vs[2][64][32];
  __shared__ float    Ss[2][64][64];
  __shared__ float    rsum[2][64];

  int lane = threadIdx.x & 31;
  int wv   = threadIdx.x >> 5;
  int win  = blockIdx.x;                 // b*81+wi : 0..1943
  int head = blockIdx.y * 2 + wv;        // 0..7
  int wi   = win % NWIN;

  // ---- stage q (pre-scaled), k, v into LDS with b128 vectors; zero-pad 49..63 ----
  const _Float16 qscale = (_Float16)0.17677669529663689f;   // HD^-0.5
  for (int r = lane; r < 64; r += 32) {
    v8h q0, q1, q2, q3, k0, k1, k2, k3, v0, v1, v2, v3;
    if (r < NTOK) {
      const _Float16* src = qkv + ((size_t)win * NTOK + r) * 768;
      const v8h* qp = (const v8h*)(src + head * HDIM);          // 16B aligned
      const v8h* kp = (const v8h*)(src + 256 + head * HDIM);
      const v8h* vp = (const v8h*)(src + 512 + head * HDIM);
      q0 = qp[0] * qscale;  q1 = qp[1] * qscale;  q2 = qp[2] * qscale;  q3 = qp[3] * qscale;
      k0 = kp[0];           k1 = kp[1];           k2 = kp[2];           k3 = kp[3];
      v0 = vp[0];           v1 = vp[1];           v2 = vp[2];           v3 = vp[3];
    } else {
      q0 = (v8h){0,0,0,0,0,0,0,0};
      q1 = q0; q2 = q0; q3 = q0; k0 = q0; k1 = q0; k2 = q0; k3 = q0; v0 = q0; v1 = q0; v2 = q0; v3 = q0;
    }
    *(v8h*)(&qs[wv][r][0]) = q0;  *(v8h*)(&qs[wv][r][8]) = q1;  *(v8h*)(&qs[wv][r][16]) = q2;  *(v8h*)(&qs[wv][r][24]) = q3;
    *(v8h*)(&ks[wv][r][0]) = k0;  *(v8h*)(&ks[wv][r][8]) = k1;  *(v8h*)(&ks[wv][r][16]) = k2;  *(v8h*)(&ks[wv][r][24]) = k3;
    *(v8h*)(&vs[wv][r][0]) = v0;  *(v8h*)(&vs[wv][r][8]) = v1;  *(v8h*)(&vs[wv][r][16]) = v2;  *(v8h*)(&vs[wv][r][24]) = v3;
  }
  __syncthreads();

  // ---- S = q k^T  (64x64 padded, K=32: one wmma per 16x16 tile) ----
  v16h bk[4];
#pragma unroll
  for (int nt = 0; nt < 4; ++nt) {
    int n = nt * 16 + (lane & 15);
#pragma unroll
    for (int h = 0; h < 16; ++h) bk[nt][h] = ks[wv][n][kform(h, lane)];
  }
#pragma unroll
  for (int mtt = 0; mtt < 4; ++mtt) {
    v16h a;
    int m = mtt * 16 + (lane & 15);
#pragma unroll
    for (int h = 0; h < 16; ++h) a[h] = qs[wv][m][kform(h, lane)];
    int mrow = mtt * 16 + ((lane & 16) ? 8 : 0);
#pragma unroll
    for (int nt = 0; nt < 4; ++nt) {
      v8f c = (v8f){0.f,0.f,0.f,0.f,0.f,0.f,0.f,0.f};
      c = wmma16(a, bk[nt], c);
      int ncol = nt * 16 + (lane & 15);
#pragma unroll
      for (int r = 0; r < 8; ++r) Ss[wv][mrow + r][ncol] = c[r];
    }
  }
  __syncthreads();

  // ---- softmax rows with relative-position bias + shift mask (computed inline) ----
  int wr7 = (wi / 9) * WS7, wc7 = (wi % 9) * WS7;
  for (int m = lane; m < 64; m += 32) {
    if (m < NTOK) {
      int iq = m / WS7, jq = m % WS7;
      int regq = sregion(wr7 + iq) * 3 + sregion(wc7 + jq);
      float mx = -1e30f;
      for (int kk = 0; kk < NTOK; ++kk) {
        int ik = kk / WS7, jk = kk % WS7;
        int regk = sregion(wr7 + ik) * 3 + sregion(wc7 + jk);
        int ridx = (iq - ik + WS7 - 1) * (2 * WS7 - 1) + (jq - jk + WS7 - 1);
        float sv = Ss[wv][m][kk] + relTab[ridx * NHEADS + head]
                 + ((regq == regk) ? 0.f : -100.f);
        Ss[wv][m][kk] = sv;
        mx = fmaxf(mx, sv);
      }
      float sum = 0.f;
      for (int kk = 0; kk < NTOK; ++kk) {
        float e = __expf(Ss[wv][m][kk] - mx);
        Ss[wv][m][kk] = e;
        sum += e;
      }
      for (int kk = NTOK; kk < 64; ++kk) Ss[wv][m][kk] = 0.f;
      rsum[wv][m] = 1.f / sum;
    } else {
      for (int kk = 0; kk < 64; ++kk) Ss[wv][m][kk] = 0.f;
      rsum[wv][m] = 0.f;
    }
  }
  __syncthreads();

  // ---- O = P v  (M=64 pad, N=32, K=64 pad -> 2 k-steps per tile) ----
  v16h bv[2][2];
#pragma unroll
  for (int kt = 0; kt < 2; ++kt)
#pragma unroll
    for (int nt = 0; nt < 2; ++nt) {
      int n = nt * 16 + (lane & 15);
#pragma unroll
      for (int h = 0; h < 16; ++h)
        bv[kt][nt][h] = vs[wv][kt * 32 + kform(h, lane)][n];
    }
#pragma unroll
  for (int mtt = 0; mtt < 4; ++mtt) {
    v8f acc[2];
    acc[0] = (v8f){0.f,0.f,0.f,0.f,0.f,0.f,0.f,0.f};
    acc[1] = acc[0];
    int m = mtt * 16 + (lane & 15);
    float rs = rsum[wv][m];
#pragma unroll
    for (int kt = 0; kt < 2; ++kt) {
      v16h a;
#pragma unroll
      for (int h = 0; h < 16; ++h)
        a[h] = (_Float16)(Ss[wv][m][kt * 32 + kform(h, lane)] * rs);
#pragma unroll
      for (int nt = 0; nt < 2; ++nt) acc[nt] = wmma16(a, bv[kt][nt], acc[nt]);
    }
    int mrow = mtt * 16 + ((lane & 16) ? 8 : 0);
#pragma unroll
    for (int nt = 0; nt < 2; ++nt) {
      int c = head * HDIM + nt * 16 + (lane & 15);
#pragma unroll
      for (int r = 0; r < 8; ++r) {
        int mm = mrow + r;
        if (mm < NTOK)
          A2[pk_idx((size_t)win * NTOK + mm, c, 8)] = (_Float16)acc[nt][r];
      }
    }
  }
}

// ---------------- launch ----------------
extern "C" void kernel_launch(void* const* d_in, const int* in_sizes, int n_in,
                              void* d_out, int out_size, void* d_ws, size_t ws_size,
                              hipStream_t stream) {
  const float* x     = (const float*)d_in[0];
  const float* g1    = (const float*)d_in[1];
  const float* b1    = (const float*)d_in[2];
  const float* w_qkv = (const float*)d_in[3];
  const float* b_qkv = (const float*)d_in[4];
  const float* relT  = (const float*)d_in[5];
  const float* w_prj = (const float*)d_in[6];
  const float* b_prj = (const float*)d_in[7];
  const float* g2    = (const float*)d_in[8];
  const float* b2    = (const float*)d_in[9];
  const float* w_fc1 = (const float*)d_in[10];
  const float* b_fc1 = (const float*)d_in[11];
  const float* w_fc2 = (const float*)d_in[12];
  const float* b_fc2 = (const float*)d_in[13];
  float* out = (float*)d_out;

  char* ws = (char*)d_ws;
  size_t offA   = 0;                               // packed-A1 / A2 (aliased) 48.8 MB
  size_t offQKV = offA   + (size_t)MP1 * 256 * 2;  // qkv f16, later reused as A3
  size_t offX1  = offQKV + (size_t)MP1 * 768 * 2;  // residual-1 f32 88.5 MB
  size_t offH   = offX1  + (size_t)MTOK * 256 * 4; // fc1 packed-A 177 MB
  size_t offWq  = offH   + (size_t)MP2 * 1024 * 2;
  size_t offWp  = offWq  + (size_t)256 * 768 * 2;
  size_t offW1  = offWp  + (size_t)256 * 256 * 2;
  size_t offW2  = offW1  + (size_t)256 * 1024 * 2;

  _Float16* A12  = (_Float16*)(ws + offA);
  _Float16* QKVb = (_Float16*)(ws + offQKV);
  _Float16* A3   = (_Float16*)(ws + offQKV);       // alias: qkv dead after attention
  float*    X1   = (float*)   (ws + offX1);
  _Float16* Hbuf = (_Float16*)(ws + offH);
  _Float16* Wq   = (_Float16*)(ws + offWq);
  _Float16* Wp   = (_Float16*)(ws + offWp);
  _Float16* W1   = (_Float16*)(ws + offW1);
  _Float16* W2   = (_Float16*)(ws + offW2);

  // 0) pack weights into WMMA B-fragment layout
  pack_w_kernel<<<(256 * 768 + 255) / 256, 256, 0, stream>>>(w_qkv, Wq, 256, 768);
  pack_w_kernel<<<(256 * 256 + 255) / 256, 256, 0, stream>>>(w_prj, Wp, 256, 256);
  pack_w_kernel<<<(256 * 1024 + 255) / 256, 256, 0, stream>>>(w_fc1, W1, 256, 1024);
  pack_w_kernel<<<(1024 * 256 + 255) / 256, 256, 0, stream>>>(w_fc2, W2, 1024, 256);

  // 1) LN1 + shift/pad/window gather -> packed-A
  ln1_window_pack<<<NWROWS / 8, dim3(32, 8), 0, stream>>>(x, g1, b1, A12);

  // 2) QKV GEMM: [MP1 x 256] x [256 x 768] -> f16 row-major
  gemm_wmma<0><<<dim3(MT1, 3), 128, 0, stream>>>(A12, Wq, b_qkv, 8,
                                                 QKVb, nullptr, nullptr, MP1);

  // 3) windowed attention -> packed-A for proj (overwrites A12)
  attn_kernel<<<dim3(BATCH * NWIN, 4), 64, 0, stream>>>(QKVb, relT, A12);

  // 4) proj GEMM + window-reverse scatter + residual -> X1
  gemm_wmma<1><<<dim3(MT1, 1), 128, 0, stream>>>(A12, Wp, b_prj, 8,
                                                 nullptr, X1, x, NWROWS);

  // 5) LN2 -> packed-A (reuses qkv region)
  ln2_pack<<<MTOK / 8, dim3(32, 8), 0, stream>>>(X1, g2, b2, A3);

  // 6) fc1 GEMM + GELU -> packed-A for fc2
  gemm_wmma<2><<<dim3(MT2, 4), 128, 0, stream>>>(A3, W1, b_fc1, 8,
                                                 Hbuf, nullptr, nullptr, MTOK);

  // 7) fc2 GEMM + residual -> d_out
  gemm_wmma<3><<<dim3(MT2, 1), 128, 0, stream>>>(Hbuf, W2, b_fc2, 64,
                                                 nullptr, out, X1, MTOK);
}
